// DecoderBlock_41609643163860
// MI455X (gfx1250) — matrix operand order repaired and checked by reading, not compiled, not run on hardware
//
#include <hip/hip_runtime.h>

typedef __bf16 bf16;
typedef __attribute__((ext_vector_type(8)))  bf16  bf16x8;
typedef __attribute__((ext_vector_type(16))) bf16  v16bf;
typedef __attribute__((ext_vector_type(8)))  float v8f;
typedef __attribute__((ext_vector_type(4)))  int   v4i;

// ---- CDNA5 async global->LDS path (guarded; falls back to VGPR staging).
// Probe-confirmed signature: (addrspace(1) v4i*, addrspace(3) v4i*, imm, imm)
#if defined(__has_builtin)
#if __has_builtin(__builtin_amdgcn_global_load_async_to_lds_b128)
#define USE_ASYNC_LDS 1
#endif
#endif

__device__ __forceinline__ void g2l_b128(const bf16* g, bf16* l) {
#ifdef USE_ASYNC_LDS
  __builtin_amdgcn_global_load_async_to_lds_b128(
      (__attribute__((address_space(1))) v4i*)g,
      (__attribute__((address_space(3))) v4i*)l, 0, 0);
#else
  *(bf16x8*)l = *(const bf16x8*)g;
#endif
}

__device__ __forceinline__ void async_wait() {
#ifdef USE_ASYNC_LDS
  asm volatile("s_wait_asynccnt 0" ::: "memory");
#endif
}

union FragU { v16bf v; bf16x8 h[2]; };

// A fragment (16x32, bf16), row-major source with element stride `stride`.
// Lane l: row = l&15 ; k-base = (l<16)?0:8 ; holds K {kb..kb+7, kb+16..kb+23}.
__device__ __forceinline__ v16bf frag_a(const bf16* base, int stride, int k0, int lane) {
  const int row = lane & 15;
  const int kb  = (lane < 16) ? 0 : 8;
  const bf16* p = base + (size_t)row * stride + k0 + kb;
  FragU u;
  u.h[0] = *(const bf16x8*)(p);
  u.h[1] = *(const bf16x8*)(p + 16);
  return u.v;
}

// B fragment (32x16, bf16) from B^T storage [N][K] with element stride `stride`.
// Lane l: col = l&15 ; k-base = (l<16)?0:8 ; holds K {kb..kb+7, kb+16..kb+23}.
__device__ __forceinline__ v16bf frag_b(const bf16* base, int stride, int k0, int lane) {
  const int col = lane & 15;
  const int kb  = (lane < 16) ? 0 : 8;
  const bf16* p = base + (size_t)col * stride + k0 + kb;
  FragU u;
  u.h[0] = *(const bf16x8*)(p);
  u.h[1] = *(const bf16x8*)(p + 16);
  return u.v;
}

__device__ __forceinline__ v8f wmma_bf16(v16bf a, v16bf b, v8f c) {
  // (neg_a, A, neg_b, B, c_mod, C, reuse_a, reuse_b)
  return __builtin_amdgcn_wmma_f32_16x16x32_bf16(false, a, false, b, (short)0, c,
                                                 false, false);
}

// ---------------------------------------------------------------------------
// GEMM: C[M,N] = A[M,K] (bf16, row-major) * B (given as B^T[N,K] bf16)
// epi: 0 = store bf16 ; 1 = +bias, ReLU, store bf16 ; 2 = +bias, store f32
// Block tile 128x256, 8 waves (2x4), wave tile 64x64 (4x4 accumulators):
// 16 WMMA per 8 LDS fragment loads per k-step.
// ---------------------------------------------------------------------------
#define GT_M 128
#define GT_N 256
#define GT_K 32
#define GSTR 40  // padded LDS stride (conflict-free b128 fragment loads)

__global__ __launch_bounds__(256) void gemm_bf16_kernel(
    const bf16* __restrict__ A, const bf16* __restrict__ Bt,
    const float* __restrict__ bias, bf16* __restrict__ Cb,
    float* __restrict__ Cf, int M, int N, int K, int epi)
{
  __shared__ __align__(16) bf16 sA[GT_M * GSTR];
  __shared__ __align__(16) bf16 sB[GT_N * GSTR];
  const int tid  = threadIdx.x;
  const int lane = tid & 31;
  const int wave = tid >> 5;
  const int wm = wave >> 2;          // 0..1  (64 rows each)
  const int wn = wave & 3;           // 0..3  (64 cols each)
  const int bm = blockIdx.x * GT_M;
  const int bn = blockIdx.y * GT_N;

  v8f acc[4][4] = {};

  const int arow = tid >> 1;         // 0..127 (A: half row / thread)
  const int acol = (tid & 1) * 16;   // 0 or 16
  const int brow = tid;              // 0..255 (B^T: full 32-elem row / thread)

  for (int kt = 0; kt < K; kt += GT_K) {
    const bf16* ga = A  + (size_t)(bm + arow) * K + kt + acol;
    g2l_b128(ga,     sA + arow * GSTR + acol);
    g2l_b128(ga + 8, sA + arow * GSTR + acol + 8);
    const bf16* gb = Bt + (size_t)(bn + brow) * K + kt;
    bf16* lb = sB + brow * GSTR;
    g2l_b128(gb,      lb);
    g2l_b128(gb + 8,  lb + 8);
    g2l_b128(gb + 16, lb + 16);
    g2l_b128(gb + 24, lb + 24);
    async_wait();
    __syncthreads();

    v16bf af[4];
#pragma unroll
    for (int i = 0; i < 4; ++i)
      af[i] = frag_a(sA + (wm * 64 + i * 16) * GSTR, GSTR, 0, lane);
#pragma unroll
    for (int j = 0; j < 4; ++j) {
      v16bf bfr = frag_b(sB + (wn * 64 + j * 16) * GSTR, GSTR, 0, lane);
#pragma unroll
      for (int i = 0; i < 4; ++i)
        acc[i][j] = wmma_bf16(af[i], bfr, acc[i][j]);
    }
    __syncthreads();
  }

  const int mofs = (lane < 16) ? 0 : 8;
#pragma unroll
  for (int i = 0; i < 4; ++i) {
#pragma unroll
    for (int j = 0; j < 4; ++j) {
      const int n  = bn + wn * 64 + j * 16 + (lane & 15);
      const int m0 = bm + wm * 64 + i * 16 + mofs;
      float bv = (epi != 0) ? bias[n] : 0.f;
#pragma unroll
      for (int r = 0; r < 8; ++r) {
        float v = acc[i][j][r] + bv;
        if (epi == 1) v = v > 0.f ? v : 0.f;
        const size_t idx = (size_t)(m0 + r) * N + n;
        if (epi == 2) Cf[idx] = v;
        else          Cb[idx] = (bf16)v;
      }
    }
  }
}

// ---------------------------------------------------------------------------
// Flash attention: per (b,h), 128 Q rows per block (one 16-row stripe/wave),
// 64-wide KV tiles, online softmax, P routed through LDS (C-layout -> A-layout).
// Q,K bf16 [B*L][D] (head = 64 cols at h*64); Vt bf16 [B*H][64][LK]; O f32.
// ---------------------------------------------------------------------------
__global__ __launch_bounds__(256) void attention_kernel(
    const bf16* __restrict__ Q, const bf16* __restrict__ Kb,
    const bf16* __restrict__ Vt, float* __restrict__ O,
    int H, int LQ, int LK, int Dm, float scale)
{
  const int hd = 64;
  const int bh = blockIdx.y;
  const int b  = bh / H;
  const int h  = bh % H;
  const int q0 = blockIdx.x * 128;
  const int tid  = threadIdx.x;
  const int lane = tid & 31;
  const int wave = tid >> 5;

  const bf16* Qp = Q  + (size_t)b * LQ * Dm + h * hd;
  const bf16* Kp = Kb + (size_t)b * LK * Dm + h * hd;
  const bf16* Vp = Vt + (size_t)bh * hd * LK;

  __shared__ __align__(16) bf16 sK[64 * 72];   // K tile   [lk=64][hd=64]
  __shared__ __align__(16) bf16 sV[64 * 72];   // V^T tile [hd=64][lk=64]
  __shared__ __align__(16) bf16 sP[128 * 72];  // P, 16 rows x 64 per wave

  v16bf qf0, qf1;
  {
    const bf16* qb = Qp + (size_t)(q0 + wave * 16) * Dm;
    qf0 = frag_a(qb, Dm, 0, lane);
    qf1 = frag_a(qb, Dm, 32, lane);
  }

  float mrun[8], lrun[8];
  v8f oacc[4] = {};
#pragma unroll
  for (int r = 0; r < 8; ++r) { mrun[r] = -3.0e38f; lrun[r] = 0.f; }

  const int lr = tid >> 2;         // 0..63
  const int lc = (tid & 3) * 16;   // 0,16,32,48

  for (int lk0 = 0; lk0 < LK; lk0 += 64) {
    const bf16* gk = Kp + (size_t)(lk0 + lr) * Dm + lc;
    const bf16* gv = Vp + (size_t)lr * LK + lk0 + lc;
    g2l_b128(gk,     sK + lr * 72 + lc);
    g2l_b128(gk + 8, sK + lr * 72 + lc + 8);
    g2l_b128(gv,     sV + lr * 72 + lc);
    g2l_b128(gv + 8, sV + lr * 72 + lc + 8);
    async_wait();
    __syncthreads();

    // S = Q * K^T (16 x 64), k-dim = hd = 64 -> 2 k-steps
    v8f s[4] = {};
#pragma unroll
    for (int t = 0; t < 4; ++t) {
      v16bf kb0 = frag_b(sK + (t * 16) * 72, 72, 0, lane);
      v16bf kb1 = frag_b(sK + (t * 16) * 72, 72, 32, lane);
      s[t] = wmma_bf16(qf0, kb0, s[t]);
      s[t] = wmma_bf16(qf1, kb1, s[t]);
    }

    // Online softmax. C-layout: VGPR slot r <-> row r (lanes 0-15) / r+8.
    bf16* pbase = sP + (size_t)(wave * 16) * 72;
    const int prow = (lane < 16) ? 0 : 8;
#pragma unroll
    for (int r = 0; r < 8; ++r) {
      float mx = -3.0e38f;
#pragma unroll
      for (int t = 0; t < 4; ++t) { s[t][r] *= scale; mx = fmaxf(mx, s[t][r]); }
#pragma unroll
      for (int off = 8; off > 0; off >>= 1) mx = fmaxf(mx, __shfl_xor(mx, off, 32));
      const float mn = fmaxf(mrun[r], mx);
      const float ef = __expf(mrun[r] - mn);
      mrun[r] = mn;
      float sum = 0.f;
#pragma unroll
      for (int t = 0; t < 4; ++t) {
        const float p = __expf(s[t][r] - mn);
        sum += p;
        pbase[(r + prow) * 72 + t * 16 + (lane & 15)] = (bf16)p;
      }
#pragma unroll
      for (int off = 8; off > 0; off >>= 1) sum += __shfl_xor(sum, off, 32);
      lrun[r] = lrun[r] * ef + sum;
      oacc[0][r] *= ef; oacc[1][r] *= ef; oacc[2][r] *= ef; oacc[3][r] *= ef;
    }

    // O += P (16x64) * V (64x64); same-wave LDS is in-order.
    v16bf pa0 = frag_a(pbase, 72, 0, lane);
    v16bf pa1 = frag_a(pbase, 72, 32, lane);
#pragma unroll
    for (int t = 0; t < 4; ++t) {
      v16bf vb0 = frag_b(sV + (t * 16) * 72, 72, 0, lane);
      v16bf vb1 = frag_b(sV + (t * 16) * 72, 72, 32, lane);
      oacc[t] = wmma_bf16(pa0, vb0, oacc[t]);
      oacc[t] = wmma_bf16(pa1, vb1, oacc[t]);
    }
    __syncthreads();
  }

  float* Op = O + ((size_t)b * LQ + q0 + wave * 16) * Dm + h * hd;
  const int mofs = (lane < 16) ? 0 : 8;
#pragma unroll
  for (int t = 0; t < 4; ++t)
#pragma unroll
    for (int r = 0; r < 8; ++r)
      Op[(size_t)(r + mofs) * Dm + t * 16 + (lane & 15)] = oacc[t][r] / lrun[r];
}

// ---------------------------------------------------------------------------
// Residual + LayerNorm over D=512 (one block/row, 256 threads, 2 elems each).
// Emits f32 (residual chain / final output) and optional bf16 (next GEMM A).
// ---------------------------------------------------------------------------
__global__ __launch_bounds__(256) void add_ln_kernel(
    const float* __restrict__ X, const float* __restrict__ R,
    const float* __restrict__ g, const float* __restrict__ bta,
    float* __restrict__ outF, bf16* __restrict__ outB, int D)
{
  const int row = blockIdx.x;
  const int tid = threadIdx.x;
  const float* x = X + (size_t)row * D;
  const float* r = R + (size_t)row * D;
  __shared__ float red[256];
  const float v0 = x[tid] + r[tid];
  const float v1 = x[tid + 256] + r[tid + 256];
  red[tid] = v0 + v1;
  __syncthreads();
  for (int o = 128; o > 0; o >>= 1) { if (tid < o) red[tid] += red[tid + o]; __syncthreads(); }
  const float mean = red[0] / (float)D;
  __syncthreads();
  const float d0 = v0 - mean, d1 = v1 - mean;
  red[tid] = d0 * d0 + d1 * d1;
  __syncthreads();
  for (int o = 128; o > 0; o >>= 1) { if (tid < o) red[tid] += red[tid + o]; __syncthreads(); }
  const float inv = rsqrtf(red[0] / (float)D + 1e-5f);
  const float y0 = d0 * inv * g[tid]       + bta[tid];
  const float y1 = d1 * inv * g[tid + 256] + bta[tid + 256];
  outF[(size_t)row * D + tid]       = y0;
  outF[(size_t)row * D + tid + 256] = y1;
  if (outB) {
    outB[(size_t)row * D + tid]       = (bf16)y0;
    outB[(size_t)row * D + tid + 256] = (bf16)y1;
  }
}

// f32 -> bf16 convert
__global__ void cvt_bf16_kernel(const float* __restrict__ in, bf16* __restrict__ out, int n) {
  const int i = blockIdx.x * 256 + threadIdx.x;
  if (i < n) out[i] = (bf16)in[i];
}

// f32 [rows][cols] -> bf16 transposed [cols][rows]
__global__ void transpose_cvt_kernel(const float* __restrict__ in, bf16* __restrict__ out,
                                     int rows, int cols) {
  const int i = blockIdx.x * 256 + threadIdx.x;
  if (i >= rows * cols) return;
  const int rr = i / cols, cc = i % cols;
  out[(size_t)cc * rows + rr] = (bf16)in[i];
}

// V bf16 [B*LK][D] -> Vt bf16 [B*H][hd][LK]
__global__ void transpose_v_kernel(const bf16* __restrict__ V, bf16* __restrict__ Vt,
                                   int H, int L, int D, int n) {
  const int i = blockIdx.x * 256 + threadIdx.x;
  if (i >= n) return;
  const int hd = D / H;
  const int dd = i % D;
  const int row = i / D;
  const int l = row % L, b = row / L;
  const int h = dd / hd, c = dd % hd;
  Vt[(((size_t)(b * H + h)) * hd + c) * L + l] = V[i];
}

// ---------------------------------------------------------------------------
extern "C" void kernel_launch(void* const* d_in, const int* in_sizes, int n_in,
                              void* d_out, int out_size, void* d_ws, size_t ws_size,
                              hipStream_t stream) {
  const int Bn = 2, LQ = 2048, LK = 2048, D = 512, H = 8, DFF = 2048;
  const int M = Bn * LQ;  // 4096

  const float* inp  = (const float*)d_in[0];
  const float* enc  = (const float*)d_in[1];
  const float* Wq   = (const float*)d_in[2];
  const float* Wk   = (const float*)d_in[3];
  const float* Wv   = (const float*)d_in[4];
  const float* ln1g = (const float*)d_in[5];
  const float* ln1b = (const float*)d_in[6];
  const float* W1   = (const float*)d_in[7];
  const float* b1   = (const float*)d_in[8];
  const float* W2   = (const float*)d_in[9];
  const float* b2   = (const float*)d_in[10];
  const float* ln2g = (const float*)d_in[11];
  const float* ln2b = (const float*)d_in[12];
  (void)in_sizes; (void)n_in; (void)out_size; (void)ws_size;

  char* w = (char*)d_ws;
  size_t off = 0;
  auto take = [&](size_t bytes) -> void* {
    void* p = w + off;
    off += (bytes + 255) & ~(size_t)255;
    return p;
  };

  bf16*  xb   = (bf16*)take((size_t)M * D * 2);       // inputs bf16
  bf16*  eb   = (bf16*)take((size_t)M * D * 2);       // encoder bf16
  bf16*  WqT  = (bf16*)take((size_t)D * D * 2);
  bf16*  WkT  = (bf16*)take((size_t)D * D * 2);
  bf16*  WvT  = (bf16*)take((size_t)D * D * 2);
  bf16*  W1T  = (bf16*)take((size_t)DFF * D * 2);     // [DFF][D]
  bf16*  W2T  = (bf16*)take((size_t)D * DFF * 2);     // [D][DFF]
  bf16*  Qb   = (bf16*)take((size_t)M * D * 2);
  bf16*  Kbuf = (bf16*)take((size_t)M * D * 2);
  bf16*  Vb   = (bf16*)take((size_t)M * D * 2);
  bf16*  Vt   = (bf16*)take((size_t)M * D * 2);       // [B*H][64][LK]
  float* Ob   = (float*)take((size_t)M * D * 4);      // attention out
  float* o1F  = (float*)take((size_t)M * D * 4);      // LN1 out f32
  bf16*  o1B  = (bf16*)take((size_t)M * D * 2);       // LN1 out bf16
  bf16*  f1   = (bf16*)take((size_t)M * DFF * 2);     // relu(ff1) bf16
  float* f2   = (float*)take((size_t)M * D * 4);      // ff2 f32

  // Stage 0: precision conversion + weight transposes (small, bandwidth-trivial)
  cvt_bf16_kernel<<<(M * D) / 256, 256, 0, stream>>>(inp, xb, M * D);
  cvt_bf16_kernel<<<(M * D) / 256, 256, 0, stream>>>(enc, eb, M * D);
  transpose_cvt_kernel<<<(D * D) / 256, 256, 0, stream>>>(Wq, WqT, D, D);
  transpose_cvt_kernel<<<(D * D) / 256, 256, 0, stream>>>(Wk, WkT, D, D);
  transpose_cvt_kernel<<<(D * D) / 256, 256, 0, stream>>>(Wv, WvT, D, D);
  transpose_cvt_kernel<<<(D * DFF) / 256, 256, 0, stream>>>(W1, W1T, D, DFF);
  transpose_cvt_kernel<<<(DFF * D) / 256, 256, 0, stream>>>(W2, W2T, DFF, D);

  // Stage 1: Q/K/V projections (WMMA bf16)
  gemm_bf16_kernel<<<dim3(M / GT_M, D / GT_N), 256, 0, stream>>>(
      xb, WqT, nullptr, Qb, nullptr, M, D, D, 0);
  gemm_bf16_kernel<<<dim3(M / GT_M, D / GT_N), 256, 0, stream>>>(
      eb, WkT, nullptr, Kbuf, nullptr, M, D, D, 0);
  gemm_bf16_kernel<<<dim3(M / GT_M, D / GT_N), 256, 0, stream>>>(
      eb, WvT, nullptr, Vb, nullptr, M, D, D, 0);
  transpose_v_kernel<<<(M * D) / 256, 256, 0, stream>>>(Vb, Vt, H, LK, D, M * D);

  // Stage 2: fused flash attention (scale = 1/sqrt(64))
  attention_kernel<<<dim3(LQ / 128, Bn * H), 256, 0, stream>>>(
      Qb, Kbuf, Vt, Ob, H, LQ, LK, D, 0.125f);

  // Stage 3: residual + LN1
  add_ln_kernel<<<M, 256, 0, stream>>>(Ob, inp, ln1g, ln1b, o1F, o1B, D);

  // Stage 4: FFN (WMMA bf16): relu(x@W1+b1) @ W2 + b2
  gemm_bf16_kernel<<<dim3(M / GT_M, DFF / GT_N), 256, 0, stream>>>(
      o1B, W1T, b1, f1, nullptr, M, DFF, D, 1);
  gemm_bf16_kernel<<<dim3(M / GT_M, D / GT_N), 256, 0, stream>>>(
      f1, W2T, b2, nullptr, f2, M, D, DFF, 2);

  // Stage 5: residual + LN2 -> final output (f32)
  add_ln_kernel<<<M, 256, 0, stream>>>(f2, o1F, ln2g, ln2b, (float*)d_out, nullptr, D);
}
